// ScaledDotProdAttention_48361331753086
// MI455X (gfx1250) — matrix operand order repaired and checked by reading, not compiled, not run on hardware
//
#include <hip/hip_runtime.h>

typedef __attribute__((ext_vector_type(16))) _Float16 v16h;
typedef __attribute__((ext_vector_type(8)))  float    v8f;
typedef _Float16 half_t;

#define BS 2
#define SL 2048
#define DM 1024
#define NH 16
#define HD 64
#define BM 128           // query rows per workgroup (32 per wave, 2 M-tiles)
#define BN 64            // kv rows per staged tile
#define WAVES 4
#define NTHREADS 128
#define LOG2E 1.44269504088896340736f
#define MASK_VAL2 (-50000.0f * LOG2E)   // mask in log2 domain

__global__ __launch_bounds__(NTHREADS)
void fa_causal_wmma(const float* __restrict__ q, const float* __restrict__ k,
                    const float* __restrict__ v, float* __restrict__ out)
{
    __shared__ half_t Ks[BN * HD];          // [n][d] row-major f16
    __shared__ half_t Vt[HD * BN];          // [d][n] transposed f16
    __shared__ half_t Ps[WAVES * 32 * BN];  // per-wave P staging (2 M-tiles)

    const int bid    = blockIdx.x;
    const int tile_m = bid & 15;            // SL/BM = 16
    const int h      = (bid >> 4) & (NH - 1);
    const int b      = bid >> 8;

    const int tid  = threadIdx.x;
    const int wave = tid >> 5;
    const int lane = tid & 31;
    const int l15  = lane & 15;
    const int hi   = lane >> 4;             // 0: lanes 0-15, 1: lanes 16-31

    const int   q0    = tile_m * BM + wave * 32;   // first of 32 rows for this wave
    const float scale = 0.125f * LOG2E;            // hd^-0.5 * log2(e): scores in log2 domain

    // ---- Q into A-fragment layout (two 16x32 f16 tiles), pre-scaled ----
    v16h aQ[2][2];                                 // [mt][chunk]
#pragma unroll
    for (int mt = 0; mt < 2; ++mt) {
        const float* qp = q + ((size_t)(b * SL + q0 + mt * 16 + l15)) * DM + h * HD;
#pragma unroll
        for (int c = 0; c < 2; ++c) {
#pragma unroll
            for (int vv = 0; vv < 8; ++vv) {
                int kk = c * 32 + hi * 16 + vv * 2;
                float2 f = *(const float2*)(qp + kk);
                aQ[mt][c][2 * vv]     = (_Float16)(f.x * scale);
                aQ[mt][c][2 * vv + 1] = (_Float16)(f.y * scale);
            }
        }
    }

    // All-ones B fragment: rowsum(P) = P x Ones via the matrix pipe
    v16h bOne;
#pragma unroll
    for (int i = 0; i < 16; ++i) bOne[i] = (_Float16)1.0f;

    float m_prev[2][8];
    v8f o_acc[2][4];
    v8f o_sum[2];                                  // running row-sum accumulator
#pragma unroll
    for (int mt = 0; mt < 2; ++mt) {
#pragma unroll
        for (int r = 0; r < 8; ++r) m_prev[mt][r] = -1e30f;
#pragma unroll
        for (int e = 0; e < 8; ++e) o_sum[mt][e] = 0.0f;
#pragma unroll
        for (int d = 0; d < 4; ++d)
#pragma unroll
            for (int e = 0; e < 8; ++e) o_acc[mt][d][e] = 0.0f;
    }

    const int t_kv_end = 2 * tile_m + 1;           // kv tiles needed by this workgroup
    for (int t_kv = 0; t_kv <= t_kv_end; ++t_kv) {
        const int kv_base = t_kv * BN;

        __syncthreads();
        // ---- cooperative stage: K -> Ks (row major), V -> Vt (transposed), fp32 -> f16 ----
#pragma unroll
        for (int i = 0; i < 8; ++i) {
            int f4  = i * NTHREADS + tid;          // 0..1023 float4 slots
            int row = f4 >> 4;                     // 16 float4 per 64-wide row
            int c4  = (f4 & 15) << 2;
            size_t goff = ((size_t)(b * SL + kv_base + row)) * DM + h * HD + c4;
            float4 kf = *(const float4*)(k + goff);
            float4 vf = *(const float4*)(v + goff);
            half_t* kp = &Ks[row * HD + c4];
            kp[0] = (_Float16)kf.x; kp[1] = (_Float16)kf.y;
            kp[2] = (_Float16)kf.z; kp[3] = (_Float16)kf.w;
            Vt[(c4 + 0) * BN + row] = (_Float16)vf.x;
            Vt[(c4 + 1) * BN + row] = (_Float16)vf.y;
            Vt[(c4 + 2) * BN + row] = (_Float16)vf.z;
            Vt[(c4 + 3) * BN + row] = (_Float16)vf.w;
        }
        __syncthreads();

        // Wave-uniform guards: skip tiles entirely above this wave's rows.
        if (kv_base > q0 + 31) continue;
        const bool need_mask = (kv_base + BN - 1 > q0);

        // ---- S = Q K^T : 4 n-tiles, B fragments shared by both M-tiles ----
        v8f s[2][4];
#pragma unroll
        for (int nt = 0; nt < 4; ++nt) {
            const v16h bK0 = *(const v16h*)&Ks[(nt * 16 + l15) * HD +  0 + hi * 16];
            const v16h bK1 = *(const v16h*)&Ks[(nt * 16 + l15) * HD + 32 + hi * 16];
#pragma unroll
            for (int mt = 0; mt < 2; ++mt) {
                v8f acc;
#pragma unroll
                for (int e = 0; e < 8; ++e) acc[e] = 0.0f;
                acc = __builtin_amdgcn_wmma_f32_16x16x32_f16(
                          false, aQ[mt][0], false, bK0, (short)0, acc, false, false);
                acc = __builtin_amdgcn_wmma_f32_16x16x32_f16(
                          false, aQ[mt][1], false, bK1, (short)0, acc, false, false);
                s[mt][nt] = acc;
            }
        }

        // ---- causal mask on diagonal tiles ----
        if (need_mask) {
#pragma unroll
            for (int mt = 0; mt < 2; ++mt)
#pragma unroll
                for (int nt = 0; nt < 4; ++nt) {
                    int col = kv_base + nt * 16 + l15;
#pragma unroll
                    for (int r = 0; r < 8; ++r) {
                        int row = q0 + mt * 16 + r + hi * 8;
                        if (col > row) s[mt][nt][r] = MASK_VAL2;
                    }
                }
        }

        // ---- online softmax in log2 domain (v_exp_f32 is exp2: no mul needed) ----
#pragma unroll
        for (int mt = 0; mt < 2; ++mt) {
#pragma unroll
            for (int r = 0; r < 8; ++r) {
                float mx = fmaxf(fmaxf(s[mt][0][r], s[mt][1][r]),
                                 fmaxf(s[mt][2][r], s[mt][3][r]));
#pragma unroll
                for (int off = 8; off >= 1; off >>= 1)
                    mx = fmaxf(mx, __shfl_xor(mx, off, 32));
                float m_new = fmaxf(m_prev[mt][r], mx);
                float alpha = __builtin_amdgcn_exp2f(m_prev[mt][r] - m_new);
                m_prev[mt][r] = m_new;
                // rescale running accumulators (row-sum rides along for free)
                o_sum[mt][r] *= alpha;
#pragma unroll
                for (int d = 0; d < 4; ++d) o_acc[mt][d][r] *= alpha;
                // P = exp2(S' - m_new)
#pragma unroll
                for (int nt = 0; nt < 4; ++nt)
                    s[mt][nt][r] = __builtin_amdgcn_exp2f(s[mt][nt][r] - m_new);
            }
        }

        // ---- P: C-layout -> LDS -> A-layout (wave-private slab) ----
        half_t* pw = &Ps[wave * 32 * BN];
#pragma unroll
        for (int mt = 0; mt < 2; ++mt)
#pragma unroll
            for (int nt = 0; nt < 4; ++nt)
#pragma unroll
                for (int r = 0; r < 8; ++r)
                    pw[(mt * 16 + r + hi * 8) * BN + nt * 16 + l15] =
                        (_Float16)s[mt][nt][r];

        asm volatile("s_wait_dscnt 0" ::: "memory");   // same-wave LDS ordering point

        v16h aP[2][2];
#pragma unroll
        for (int mt = 0; mt < 2; ++mt)
#pragma unroll
            for (int c = 0; c < 2; ++c)
#pragma unroll
                for (int vv = 0; vv < 8; ++vv) {
                    int nn = c * 32 + hi * 16 + vv * 2;
                    aP[mt][c][2 * vv]     = pw[(mt * 16 + l15) * BN + nn];
                    aP[mt][c][2 * vv + 1] = pw[(mt * 16 + l15) * BN + nn + 1];
                }

        // ---- row sums on the matrix pipe: o_sum += P x Ones ----
#pragma unroll
        for (int mt = 0; mt < 2; ++mt) {
            o_sum[mt] = __builtin_amdgcn_wmma_f32_16x16x32_f16(
                            false, aP[mt][0], false, bOne, (short)0, o_sum[mt],
                            false, false);
            o_sum[mt] = __builtin_amdgcn_wmma_f32_16x16x32_f16(
                            false, aP[mt][1], false, bOne, (short)0, o_sum[mt],
                            false, false);
        }

        // ---- O += P V : 4 d-tiles, B fragments shared by both M-tiles ----
#pragma unroll
        for (int d = 0; d < 4; ++d) {
            const v16h bV0 = *(const v16h*)&Vt[(d * 16 + l15) * BN +  0 + hi * 16];
            const v16h bV1 = *(const v16h*)&Vt[(d * 16 + l15) * BN + 32 + hi * 16];
#pragma unroll
            for (int mt = 0; mt < 2; ++mt) {
                o_acc[mt][d] = __builtin_amdgcn_wmma_f32_16x16x32_f16(
                                   false, aP[mt][0], false, bV0, (short)0,
                                   o_acc[mt][d], false, false);
                o_acc[mt][d] = __builtin_amdgcn_wmma_f32_16x16x32_f16(
                                   false, aP[mt][1], false, bV1, (short)0,
                                   o_acc[mt][d], false, false);
            }
        }
    }

    // ---- normalize by row sums (one rcp per row), write out ----
#pragma unroll
    for (int mt = 0; mt < 2; ++mt)
#pragma unroll
        for (int r = 0; r < 8; ++r) {
            float inv = 1.0f / o_sum[mt][r];
            int row = q0 + mt * 16 + r + hi * 8;
            float* op = out + ((size_t)(b * SL + row)) * DM + h * HD;
#pragma unroll
            for (int d = 0; d < 4; ++d)
                op[d * 16 + l15] = o_acc[mt][d][r] * inv;
        }
}

extern "C" void kernel_launch(void* const* d_in, const int* in_sizes, int n_in,
                              void* d_out, int out_size, void* d_ws, size_t ws_size,
                              hipStream_t stream) {
    const float* q = (const float*)d_in[0];
    const float* k = (const float*)d_in[1];
    const float* v = (const float*)d_in[2];
    float* out = (float*)d_out;
    dim3 grid(BS * NH * (SL / BM));   // 512 workgroups
    fa_causal_wmma<<<grid, NTHREADS, 0, stream>>>(q, k, v, out);
}
